// SwapNet_2894807957828
// MI455X (gfx1250) — matrix operand order repaired and checked by reading, not compiled, not run on hardware
//
#include <hip/hip_runtime.h>
#include <cstdint>
#include <cstddef>

// ---------------------------------------------------------------------------
// CDNA5 (gfx1250) WMMA types
// ---------------------------------------------------------------------------
typedef __attribute__((ext_vector_type(16))) __bf16 v16bf;
typedef __attribute__((ext_vector_type(8)))  __bf16 v8bf;
typedef __attribute__((ext_vector_type(8)))  float  v8f;

#define WMMA_BF16(A_, B_, C_) \
  __builtin_amdgcn_wmma_f32_16x16x32_bf16(false, (A_), false, (B_), (short)0, (C_), false, false)

// A fragment: 16x32 bf16 (MxK). Lane l: row = l&15, K halves {koff..koff+7, 16+koff..}
// with koff = (l>>4)*8  (per CDNA5 ISA 16-bit A layout).
__device__ __forceinline__ v16bf load_A_frag(const __bf16* A, int lda, int mbase,
                                             int kbase, int lane) {
  int r    = mbase + (lane & 15);
  int koff = (lane >> 4) << 3;
  const __bf16* pr = A + (long)r * lda + kbase + koff;
  v8bf lo = *(const v8bf*)pr;          // K = kbase+koff .. +7   (16B load)
  v8bf hi = *(const v8bf*)(pr + 16);   // K = kbase+16+koff .. +7
  v16bf f;
#pragma unroll
  for (int i = 0; i < 8; ++i) { f[i] = lo[i]; f[i + 8] = hi[i]; }
  return f;
}

// B fragment: 32x16 bf16 (KxN). Lane l: col = l&15, K halves {koff..koff+7, 16+koff..}
// with koff = (l>>4)*8 (same K order as the A fragment).
__device__ __forceinline__ v16bf load_B_frag(const __bf16* Bm, int ldb, int kbase,
                                             int nbase, int lane) {
  const __bf16* pr = Bm + (long)(kbase + ((lane >> 4) << 3)) * ldb + nbase + (lane & 15);
  v16bf f;
#pragma unroll
  for (int i = 0; i < 8; ++i) { f[i] = pr[(long)i * ldb]; f[i + 8] = pr[(long)(16 + i) * ldb]; }
  return f;
}

// ---------------------------------------------------------------------------
// Conversion / concat kernels (f32 -> bf16 staging in workspace)
// ---------------------------------------------------------------------------
__global__ void cvt_f32_bf16_kernel(const float* __restrict__ src,
                                    __bf16* __restrict__ dst, long n) {
  long i = (long)blockIdx.x * blockDim.x + threadIdx.x;
  if (i < n) dst[i] = (__bf16)src[i];
}

// dst[b, r, c] = r < rowsA ? A[b,r,c] : (r < rowsValid ? B[b,r-rowsA,c] : 0)
__global__ void concat_rows_bf16_kernel(const float* __restrict__ A,
                                        const float* __restrict__ B,
                                        __bf16* __restrict__ dst,
                                        int rowsA, int rowsValid, int rowsPad,
                                        int cols, int nb) {
  long i = (long)blockIdx.x * blockDim.x + threadIdx.x;
  long total = (long)nb * rowsPad * cols;
  if (i >= total) return;
  int  c = (int)(i % cols);
  long t = i / cols;
  int  r = (int)(t % rowsPad);
  int  z = (int)(t / rowsPad);
  float v = 0.0f;
  if (r < rowsA)          v = A[((long)z * rowsA + r) * cols + c];
  else if (r < rowsValid) v = B[((long)z * (rowsValid - rowsA) + (r - rowsA)) * cols + c];
  dst[i] = (__bf16)v;
}

// ---------------------------------------------------------------------------
// Generic bf16 GEMM via V_WMMA_F32_16X16X32_BF16.
//   out[b, m, n] = sum_k A[b, m, k] * W[k, n] (+ bias[n])
// One wave computes a 32x64 tile: 2 A fragments x 4 B fragments -> 8 WMMAs
// per 32-wide K step (B fragments reused across both M halves).  All four B
// fragments are loaded before the WMMA burst so the scheduler can overlap
// load latency with matrix issue using partial s_wait_loadcnt.
// M must be a multiple of 32.
// mode 0: store [B, M, N] (bf16 and/or f32).  mode 1: store K^T as
// [B, N, MpadT] (bf16) so attention QK^T B-fragments are contiguous.
// ---------------------------------------------------------------------------
__global__ __launch_bounds__(256) void gemm_bf16_wmma(
    const __bf16* __restrict__ A, const __bf16* __restrict__ W,
    const float* __restrict__ bias, __bf16* __restrict__ outBf,
    float* __restrict__ outF32, int M, int N, int K,
    long strideA, long strideOut, int mode, int MpadT) {
  int lane = threadIdx.x & 31;
  int wave = threadIdx.x >> 5;
  int nT   = N >> 6;                 // 64-wide N tiles (N is a multiple of 64)
  int mT   = M >> 5;                 // 32-row M tiles
  int tile = blockIdx.x * 8 + wave;
  if (tile >= mT * nT) return;
  int mb = (tile / nT) << 5;
  int nb = (tile % nT) << 6;
  const __bf16* Ab = A + (long)blockIdx.z * strideA;

  v8f acc[2][4] = {};
  for (int kb = 0; kb < K; kb += 32) {
    // Speculative prefetch of the next weight K-block (global_prefetch_b8);
    // unconditional: OOB speculative prefetches are silently dropped.
    __builtin_prefetch((const void*)(W + (long)(kb + 32 + lane) * N + nb), 0, 1);

    v16bf a0 = load_A_frag(Ab, K, mb,      kb, lane);
    v16bf a1 = load_A_frag(Ab, K, mb + 16, kb, lane);
    v16bf b0 = load_B_frag(W, N, kb, nb,      lane);
    v16bf b1 = load_B_frag(W, N, kb, nb + 16, lane);
    v16bf b2 = load_B_frag(W, N, kb, nb + 32, lane);
    v16bf b3 = load_B_frag(W, N, kb, nb + 48, lane);

    acc[0][0] = WMMA_BF16(a0, b0, acc[0][0]);
    acc[1][0] = WMMA_BF16(a1, b0, acc[1][0]);
    acc[0][1] = WMMA_BF16(a0, b1, acc[0][1]);
    acc[1][1] = WMMA_BF16(a1, b1, acc[1][1]);
    acc[0][2] = WMMA_BF16(a0, b2, acc[0][2]);
    acc[1][2] = WMMA_BF16(a1, b2, acc[1][2]);
    acc[0][3] = WMMA_BF16(a0, b3, acc[0][3]);
    acc[1][3] = WMMA_BF16(a1, b3, acc[1][3]);
  }

  int nl = lane & 15, hi = lane >> 4;
#pragma unroll
  for (int mi = 0; mi < 2; ++mi) {
#pragma unroll
    for (int j = 0; j < 4; ++j) {
      int n = nb + j * 16 + nl;
      float bv = bias ? bias[n] : 0.0f;
#pragma unroll
      for (int v = 0; v < 8; ++v) {
        int m = mb + mi * 16 + hi * 8 + v;   // C/D layout: c[v] <-> (m, n)
        float val = acc[mi][j][v] + bv;
        if (mode == 0) {
          long o = (long)blockIdx.z * strideOut + (long)m * N + n;
          if (outBf)  outBf[o]  = (__bf16)val;
          if (outF32) outF32[o] = val;
        } else {
          outBf[((long)blockIdx.z * N + n) * (long)MpadT + m] = (__bf16)val;
        }
      }
    }
  }
}

// ---------------------------------------------------------------------------
// Flash attention (per-wave 16-query tile, online softmax, all-WMMA).
//   Q:  [B, Sq, C] bf16   (head h occupies cols h*64..h*64+63)
//   KT: [B, C, SkvPad] bf16 (transposed K projection)
//   V:  [B, SkvPad, C] bf16 (padding rows are zero)
//   Out:[B, Sq, C] bf16
// ---------------------------------------------------------------------------
#define NEG_BIG (-3.0e38f)

__global__ __launch_bounds__(256) void flash_attn_wmma(
    const __bf16* __restrict__ Q, const __bf16* __restrict__ KT,
    const __bf16* __restrict__ V, __bf16* __restrict__ Out,
    int Bn, int H, int Sq, int SkvPad, int SkvValid, int C, float scale) {
  __shared__ __attribute__((aligned(16))) __bf16 Plds[8][16 * 32];

  int lane = threadIdx.x & 31;
  int wave = threadIdx.x >> 5;
  int qT = Sq >> 4;
  int id = blockIdx.x * 8 + wave;
  if (id >= Bn * H * qT) return;
  int qt = id % qT;
  int bh = id / qT;
  int h = bh % H, b = bh / H;
  int d0 = h * 64;

  const __bf16* Qb  = Q  + (long)b * Sq * C + d0;
  const __bf16* KTb = KT + ((long)b * C + d0) * SkvPad;
  const __bf16* Vb  = V  + (long)b * SkvPad * C + d0;
  int mb = qt * 16;
  int nl = lane & 15, hi = lane >> 4;

  // Q fragments held in registers for the whole key loop (d = 64 -> 2 frags).
  v16bf qa0 = load_A_frag(Qb, C, mb, 0,  lane);
  v16bf qa1 = load_A_frag(Qb, C, mb, 32, lane);

  float mr[8], sr[8];
#pragma unroll
  for (int v = 0; v < 8; ++v) { mr[v] = NEG_BIG; sr[v] = 0.0f; }
  v8f o0 = {}, o1 = {}, o2 = {}, o3 = {};

  for (int kt = 0; kt < SkvPad; kt += 32) {
    // speculative prefetch of next key tile (K^T stream + V stream)
    __builtin_prefetch((const void*)(KTb + (long)lane * SkvPad + kt + 32), 0, 1);
    __builtin_prefetch((const void*)(Vb + (long)(kt + 32 + lane) * C), 0, 1);

    // scores S = Q K^T for 32 keys (two 16-key C tiles)
    v16bf k00 = load_B_frag(KTb, SkvPad, 0,  kt,      lane);
    v16bf k01 = load_B_frag(KTb, SkvPad, 32, kt,      lane);
    v16bf k10 = load_B_frag(KTb, SkvPad, 0,  kt + 16, lane);
    v16bf k11 = load_B_frag(KTb, SkvPad, 32, kt + 16, lane);
    v8f s0 = {}, s1 = {};
    s0 = WMMA_BF16(qa0, k00, s0);
    s0 = WMMA_BF16(qa1, k01, s0);
    s1 = WMMA_BF16(qa0, k10, s1);
    s1 = WMMA_BF16(qa1, k11, s1);

    bool ok0 = (kt + nl)      < SkvValid;
    bool ok1 = (kt + 16 + nl) < SkvValid;
#pragma unroll
    for (int v = 0; v < 8; ++v) {
      float a0 = ok0 ? s0[v] * scale : NEG_BIG;
      float a1 = ok1 ? s1[v] * scale : NEG_BIG;
      // row max across the 16 lanes of this half (one attention row per half;
      // xor masks 1/2/4/8 never cross the 16-lane boundary)
      float t = fmaxf(a0, a1);
      t = fmaxf(t, __shfl_xor(t, 1));
      t = fmaxf(t, __shfl_xor(t, 2));
      t = fmaxf(t, __shfl_xor(t, 4));
      t = fmaxf(t, __shfl_xor(t, 8));
      float mnew = fmaxf(mr[v], t);
      float corr = __expf(mr[v] - mnew);
      float p0 = __expf(a0 - mnew);
      float p1 = __expf(a1 - mnew);
      if (!ok0) p0 = 0.0f;
      if (!ok1) p1 = 0.0f;
      float ps = p0 + p1;
      ps += __shfl_xor(ps, 1);
      ps += __shfl_xor(ps, 2);
      ps += __shfl_xor(ps, 4);
      ps += __shfl_xor(ps, 8);
      sr[v] = sr[v] * corr + ps;
      mr[v] = mnew;
      o0[v] *= corr; o1[v] *= corr; o2[v] *= corr; o3[v] *= corr;
      // stage P in LDS (C layout) so we can reload it in A-fragment layout
      int row = hi * 8 + v;
      Plds[wave][row * 32 + nl]      = (__bf16)p0;
      Plds[wave][row * 32 + 16 + nl] = (__bf16)p1;
    }
    asm volatile("s_wait_dscnt 0" ::: "memory");  // per-wave LDS RAW fence

    v16bf pa = load_A_frag(&Plds[wave][0], 32, 0, 0, lane);  // P as 16x32 A
    // O += P V  (d = 64 -> 4 output tiles of 16 cols; V rows contiguous)
    v16bf vb0 = load_B_frag(Vb, C, kt, 0,  lane);
    v16bf vb1 = load_B_frag(Vb, C, kt, 16, lane);
    v16bf vb2 = load_B_frag(Vb, C, kt, 32, lane);
    v16bf vb3 = load_B_frag(Vb, C, kt, 48, lane);
    o0 = WMMA_BF16(pa, vb0, o0);
    o1 = WMMA_BF16(pa, vb1, o1);
    o2 = WMMA_BF16(pa, vb2, o2);
    o3 = WMMA_BF16(pa, vb3, o3);
  }

  __bf16* Ob = Out + (long)b * Sq * C + d0;
#pragma unroll
  for (int v = 0; v < 8; ++v) {
    int m = mb + hi * 8 + v;
    float inv = 1.0f / sr[v];
    Ob[(long)m * C + nl]      = (__bf16)(o0[v] * inv);
    Ob[(long)m * C + 16 + nl] = (__bf16)(o1[v] * inv);
    Ob[(long)m * C + 32 + nl] = (__bf16)(o2[v] * inv);
    Ob[(long)m * C + 48 + nl] = (__bf16)(o3[v] * inv);
  }
}

// ---------------------------------------------------------------------------
// Host-side orchestration
// ---------------------------------------------------------------------------
extern "C" void kernel_launch(void* const* d_in, const int* in_sizes, int n_in,
                              void* d_out, int out_size, void* d_ws, size_t ws_size,
                              hipStream_t stream) {
  (void)in_sizes; (void)n_in; (void)out_size; (void)ws_size;

  const float* hidden  = (const float*)d_in[0];
  const float* facenet = (const float*)d_in[1];
  const float* enc     = (const float*)d_in[2];
  const float* idf     = (const float*)d_in[3];
  const float* Wq1 = (const float*)d_in[4];
  const float* Wk1 = (const float*)d_in[5];
  const float* Wv1 = (const float*)d_in[6];
  const float* Wo1 = (const float*)d_in[7];
  const float* bo1 = (const float*)d_in[8];
  const float* Wq2 = (const float*)d_in[9];
  const float* Wk2 = (const float*)d_in[10];
  const float* Wv2 = (const float*)d_in[11];
  const float* Wo2 = (const float*)d_in[12];
  const float* bo2 = (const float*)d_in[13];

  const int B = 2, S = 2048, C = 640, H = 10, Ce = 1024;
  const int Skv1 = 4096;                 // 2S concat for attn1 KV
  const int Skv2v = 154, Skv2p = 160;    // 77+77 padded to a 32 multiple
  const float scale = 0.125f;            // 1/sqrt(64)

  // workspace bump allocator (256B aligned)
  char* wp = (char*)d_ws;
  auto alloc = [&](size_t bytes) -> void* {
    void* r = wp; wp += (bytes + 255) & ~(size_t)255; return r;
  };
  __bf16* face_bf    = (__bf16*)alloc((size_t)B * S * C * 2);
  __bf16* concat1_bf = (__bf16*)alloc((size_t)B * Skv1 * C * 2);
  __bf16* kv2_bf     = (__bf16*)alloc((size_t)B * Skv2p * Ce * 2);
  __bf16* wq1_bf = (__bf16*)alloc((size_t)C * C * 2);
  __bf16* wk1_bf = (__bf16*)alloc((size_t)C * C * 2);
  __bf16* wv1_bf = (__bf16*)alloc((size_t)C * C * 2);
  __bf16* wo1_bf = (__bf16*)alloc((size_t)C * C * 2);
  __bf16* wq2_bf = (__bf16*)alloc((size_t)C * C * 2);
  __bf16* wk2_bf = (__bf16*)alloc((size_t)Ce * C * 2);
  __bf16* wv2_bf = (__bf16*)alloc((size_t)Ce * C * 2);
  __bf16* wo2_bf = (__bf16*)alloc((size_t)C * C * 2);
  __bf16* q1_bf   = (__bf16*)alloc((size_t)B * S * C * 2);
  __bf16* k1T_bf  = (__bf16*)alloc((size_t)B * C * Skv1 * 2);
  __bf16* v1_bf   = (__bf16*)alloc((size_t)B * Skv1 * C * 2);
  __bf16* ctx1_bf = (__bf16*)alloc((size_t)B * S * C * 2);
  __bf16* att1_bf = (__bf16*)alloc((size_t)B * S * C * 2);
  __bf16* q2_bf   = (__bf16*)alloc((size_t)B * S * C * 2);
  __bf16* k2T_bf  = (__bf16*)alloc((size_t)B * C * Skv2p * 2);
  __bf16* v2_bf   = (__bf16*)alloc((size_t)B * Skv2p * C * 2);
  __bf16* ctx2_bf = (__bf16*)alloc((size_t)B * S * C * 2);

  auto cvt = [&](const float* s, __bf16* d, long n) {
    cvt_f32_bf16_kernel<<<dim3((unsigned)((n + 255) / 256)), 256, 0, stream>>>(s, d, n);
  };

  // stage bf16 activations + weights
  cvt(facenet, face_bf, (long)B * S * C);
  {
    long n = (long)B * Skv1 * C;
    concat_rows_bf16_kernel<<<dim3((unsigned)((n + 255) / 256)), 256, 0, stream>>>(
        facenet, hidden, concat1_bf, S, 2 * S, Skv1, C, B);
  }
  {
    long n = (long)B * Skv2p * Ce;
    concat_rows_bf16_kernel<<<dim3((unsigned)((n + 255) / 256)), 256, 0, stream>>>(
        enc, idf, kv2_bf, 77, Skv2v, Skv2p, Ce, B);
  }
  cvt(Wq1, wq1_bf, (long)C * C);   cvt(Wk1, wk1_bf, (long)C * C);
  cvt(Wv1, wv1_bf, (long)C * C);   cvt(Wo1, wo1_bf, (long)C * C);
  cvt(Wq2, wq2_bf, (long)C * C);   cvt(Wo2, wo2_bf, (long)C * C);
  cvt(Wk2, wk2_bf, (long)Ce * C);  cvt(Wv2, wv2_bf, (long)Ce * C);

  auto gemm = [&](const __bf16* A, const __bf16* W, const float* bias,
                  __bf16* obf, float* of32, int M, int N, int K,
                  long sA, long sO, int mode, int MpadT) {
    int tiles = (M / 32) * (N / 64);
    gemm_bf16_wmma<<<dim3((unsigned)((tiles + 7) / 8), 1, B), 256, 0, stream>>>(
        A, W, bias, obf, of32, M, N, K, sA, sO, mode, MpadT);
  };

  // ---- attn1 ----
  gemm(face_bf,    wq1_bf, nullptr, q1_bf,  nullptr, S,    C, C, (long)S * C,    (long)S * C,    0, 0);
  gemm(concat1_bf, wk1_bf, nullptr, k1T_bf, nullptr, Skv1, C, C, (long)Skv1 * C, 0,              1, Skv1);
  gemm(concat1_bf, wv1_bf, nullptr, v1_bf,  nullptr, Skv1, C, C, (long)Skv1 * C, (long)Skv1 * C, 0, 0);
  {
    int waves = B * H * (S / 16);
    flash_attn_wmma<<<dim3((unsigned)((waves + 7) / 8)), 256, 0, stream>>>(
        q1_bf, k1T_bf, v1_bf, ctx1_bf, B, H, S, Skv1, Skv1, C, scale);
  }
  gemm(ctx1_bf, wo1_bf, bo1, att1_bf, nullptr, S, C, C, (long)S * C, (long)S * C, 0, 0);

  // ---- attn2 ----
  gemm(att1_bf, wq2_bf, nullptr, q2_bf,  nullptr, S,     C, C,  (long)S * C,      (long)S * C,     0, 0);
  gemm(kv2_bf,  wk2_bf, nullptr, k2T_bf, nullptr, Skv2p, C, Ce, (long)Skv2p * Ce, 0,               1, Skv2p);
  gemm(kv2_bf,  wv2_bf, nullptr, v2_bf,  nullptr, Skv2p, C, Ce, (long)Skv2p * Ce, (long)Skv2p * C, 0, 0);
  {
    int waves = B * H * (S / 16);
    flash_attn_wmma<<<dim3((unsigned)((waves + 7) / 8)), 256, 0, stream>>>(
        q2_bf, k2T_bf, v2_bf, ctx2_bf, B, H, S, Skv2p, Skv2v, C, scale);
  }
  gemm(ctx2_bf, wo2_bf, bo2, nullptr, (float*)d_out, S, C, C, (long)S * C, (long)S * C, 0, 0);
}
